// RelationAttention_76922864271984
// MI455X (gfx1250) — matrix operand order repaired and checked by reading, not compiled, not run on hardware
//
#include <hip/hip_runtime.h>
#include <hip/hip_bf16.h>

// Problem constants (match reference)
#define IN_DIM   12544
#define BATCH    8
#define NROWS    1024           // N (keys/values rows)
#define MROWS    1024           // M (query rows)
#define QK_DIM   1024           // APP_MDIM
#define V_DIM    256            // APP_DIM
#define HEADS    4
#define MD       256            // per-head Q/K dim (also the score scale)
#define DV       64             // per-head V dim

typedef __attribute__((ext_vector_type(16))) _Float16 v16h;
typedef __attribute__((ext_vector_type(8)))  _Float16 v8h;
typedef __attribute__((ext_vector_type(8)))  float    v8f;
typedef __attribute__((ext_vector_type(4)))  float    v4f;
typedef __attribute__((ext_vector_type(4)))  _Float16 v4h;

__device__ __forceinline__ v8f wmma_f16(v16h a, v16h b, v8f c) {
    // (neg_a, A, neg_b, B, c_mod, C, reuse_a, reuse_b)
    return __builtin_amdgcn_wmma_f32_16x16x32_f16(false, a, false, b, (short)0, c,
                                                  false, false);
}

// CDNA5 async global -> LDS copy, 16B per lane (ISA: GLOBAL_LOAD_ASYNC_TO_LDS_B128).
// lds_off = per-lane LDS byte offset (low 32 bits of the flat shared address).
__device__ __forceinline__ void async_b128(unsigned lds_off, const void* gptr) {
    asm volatile("global_load_async_to_lds_b128 %0, %1, off"
                 :
                 : "v"(lds_off), "v"((unsigned long long)(uintptr_t)gptr)
                 : "memory");
}

// ---- fragment loaders (contiguous k order k = 16*(lane/16) + element, same in A and B) ----------------

// A matrix 16x32 f16, row-major source with leading dim `ld` (elements).
// lane<16: row=lane, K=kh..kh+15 with kh=0; lane>=16: kh=16.
__device__ __forceinline__ v16h load_a_frag(const _Float16* base, int ld) {
    const int lane = threadIdx.x & 31;
    const int row  = lane & 15;
    const int kh   = (lane >> 4) << 4;        // 0 or 16
    const _Float16* p = base + (size_t)row * ld + kh;
    v8h lo = *(const v8h*)(p);                // K = kh .. kh+7
    v8h hi = *(const v8h*)(p + 8);            // K = kh+8 .. kh+15
    v16h a;
#pragma unroll
    for (int i = 0; i < 8; ++i) { a[i] = lo[i]; a[i + 8] = hi[i]; }
    return a;
}

// B matrix 32x16 f16 from "transposed" storage: for column c the 32 K-values
// are contiguous with stride `ld` between columns.
// lane<16: col=lane, K=0..15; lane>=16: col=lane-16, K=16..31.
__device__ __forceinline__ v16h load_bT_frag(const _Float16* base, int ld) {
    const int lane = threadIdx.x & 31;
    const int col  = lane & 15;
    const int kq   = (lane >> 4) << 4;        // 0 or 16
    return *(const v16h*)(base + (size_t)col * ld + kq);
}

// Same as above but from an LDS tile with fixed ld=32, split into 16B loads.
__device__ __forceinline__ v16h load_bT_frag_lds(const _Float16* base) {
    const int lane = threadIdx.x & 31;
    const int col  = lane & 15;
    const int kq   = (lane >> 4) << 4;
    const _Float16* p = base + col * 32 + kq;
    v8h lo = *(const v8h*)(p);
    v8h hi = *(const v8h*)(p + 8);
    v16h b;
#pragma unroll
    for (int i = 0; i < 8; ++i) { b[i] = lo[i]; b[i + 8] = hi[i]; }
    return b;
}

// B matrix 32x16 f16 from a row-major [K, cols] source (strided gather).
__device__ __forceinline__ v16h load_b_rowmajor_frag(const _Float16* base, int ld) {
    const int lane = threadIdx.x & 31;
    const int col  = lane & 15;
    const int kq   = (lane >> 4) << 4;
    v16h b;
#pragma unroll
    for (int e = 0; e < 16; ++e) b[e] = base[(size_t)(kq + e) * ld + col];
    return b;
}

// ---- kernels ----------------------------------------------------------------

// fp32 -> f16, 4-wide grid-stride
__global__ void cvt_f32_f16(const float* __restrict__ src,
                            _Float16* __restrict__ dst, size_t n4) {
    size_t i = (size_t)blockIdx.x * blockDim.x + threadIdx.x;
    size_t stride = (size_t)gridDim.x * blockDim.x;
    for (; i < n4; i += stride) {
        v4f s = *(const v4f*)(src + i * 4);
        v4h d;
#pragma unroll
        for (int j = 0; j < 4; ++j) d[j] = (_Float16)s[j];
        *(v4h*)(dst + i * 4) = d;
    }
}

// W [rows, cols] fp32 -> WT [cols, rows] f16 (LDS tiled transpose)
__global__ __launch_bounds__(256) void transpose_cvt(const float* __restrict__ W,
                                                     _Float16* __restrict__ WT,
                                                     int rows, int cols) {
    __shared__ float tile[32][33];
    const int tx = threadIdx.x & 31;
    const int ty = threadIdx.x >> 5;          // 0..7
    const int c0 = blockIdx.x * 32;
    const int r0 = blockIdx.y * 32;
#pragma unroll
    for (int i = 0; i < 32; i += 8) {
        int r = r0 + ty + i, c = c0 + tx;
        tile[ty + i][tx] = (r < rows && c < cols) ? W[(size_t)r * cols + c] : 0.f;
    }
    __syncthreads();
#pragma unroll
    for (int i = 0; i < 32; i += 8) {
        int c = c0 + ty + i, r = r0 + tx;
        if (c < cols && r < rows)
            WT[(size_t)c * rows + r] = (_Float16)tile[tx][ty + i];
    }
}

// out[rows, cols] = A[rows, IN_DIM] @ W + bias, W given as WT[cols, IN_DIM].
// Block = 8 waves covering 128 rows x 128 cols. All waves share the same
// column range, so the B tile (128x32 f16 = 8KB) is staged in LDS once per
// block with double-buffered async global->LDS copies. B-fragment LDS reads
// are software-pipelined (1-deep rotation) so each WMMA only needs a partial
// DScnt wait instead of a full drain.
__global__ __launch_bounds__(256) void gemm_proj(const _Float16* __restrict__ A,
                                                 const _Float16* __restrict__ WT,
                                                 const float* __restrict__ bias,
                                                 _Float16* __restrict__ out,
                                                 int cols) {
    __shared__ __align__(16) _Float16 bLds[2][128 * 32];   // 2 x 8KB
    const int tid  = threadIdx.x;
    const int wave = tid >> 5;
    const int lane = tid & 31;
    const int m0 = blockIdx.x * 128 + wave * 16;
    const int n0 = blockIdx.y * 128;

    // async-copy assignment: thread -> (column, 32B half of that column's K run)
    const int ccol  = tid >> 1;               // 0..127
    const int chalf = tid & 1;                // 0/1 -> K 0..15 / 16..31
    const _Float16* gsrc0 = WT + (size_t)(n0 + ccol) * IN_DIM + chalf * 16;
    const unsigned ldsDst0 =
        (unsigned)(uintptr_t)&bLds[0][ccol * 32 + chalf * 16];
    const unsigned ldsDst1 =
        (unsigned)(uintptr_t)&bLds[1][ccol * 32 + chalf * 16];

    v8f acc[8];
#pragma unroll
    for (int t = 0; t < 8; ++t)
#pragma unroll
        for (int r = 0; r < 8; ++r) acc[t][r] = 0.0f;

    // prologue: stage k=0 into buffer 0 (2 async ops per wave, 16B per lane)
    async_b128(ldsDst0,      gsrc0);
    async_b128(ldsDst0 + 16, gsrc0 + 8);

    const _Float16* aBase = A + (size_t)m0 * IN_DIM;
    for (int k0 = 0; k0 < IN_DIM; k0 += 32) {
        const int cur = (k0 >> 5) & 1;
        if (k0 + 32 < IN_DIM) {
            const _Float16* g = gsrc0 + (k0 + 32);
            const unsigned dst = cur ? ldsDst0 : ldsDst1;   // write buf cur^1
            async_b128(dst,      g);
            async_b128(dst + 16, g + 8);
            // 2 just-issued ops may remain in flight; current buffer drained.
            asm volatile("s_wait_asynccnt 0x2" ::: "memory");
        } else {
            asm volatile("s_wait_asynccnt 0x0" ::: "memory");
        }
        // A fragment (per-wave rows) straight from global, overlaps the wait
        v16h a = load_a_frag(aBase + k0, IN_DIM);
        __syncthreads();                       // all waves' copies landed

        // pipelined B-fragment reads: frag t+1 in flight while WMMA t issues
        v16h bcur = load_bT_frag_lds(&bLds[cur][0]);
#pragma unroll
        for (int t = 0; t < 8; ++t) {
            v16h bnext;
            if (t < 7) bnext = load_bT_frag_lds(&bLds[cur][(t + 1) * 16 * 32]);
            acc[t] = wmma_f16(a, bcur, acc[t]);
            bcur = bnext;
        }
        __syncthreads();                       // reads retired before overwrite
    }

    // C/D layout: VGPR r -> row (r + 8*(lane>>4)), col (lane&15)
    const int cl = lane & 15;
    const int rh = (lane >> 4) << 3;
#pragma unroll
    for (int t = 0; t < 8; ++t) {
        int c = n0 + t * 16 + cl;
        float bv = bias[c];
#pragma unroll
        for (int r = 0; r < 8; ++r) {
            int m = m0 + r + rh;
            out[(size_t)m * cols + c] = (_Float16)(acc[t][r] + bv);
        }
    }
}

// L2-normalize Q per (row, head): one wave per (row, head)
__global__ __launch_bounds__(256) void qnorm(const _Float16* __restrict__ Qp,
                                             _Float16* __restrict__ Qn) {
    const int wid  = (int)((blockIdx.x * blockDim.x + threadIdx.x) >> 5);
    const int lane = threadIdx.x & 31;
    const int head = wid & (HEADS - 1);
    const int row  = wid >> 2;
    const _Float16* p = Qp + (size_t)row * QK_DIM + head * MD;
    float v[8], s = 0.f;
#pragma unroll
    for (int i = 0; i < 8; ++i) { v[i] = (float)p[lane + i * 32]; s += v[i] * v[i]; }
#pragma unroll
    for (int m = 16; m >= 1; m >>= 1) s += __shfl_xor(s, m, 32);
    float inv = rsqrtf(s);
    _Float16* q = Qn + (size_t)row * QK_DIM + head * MD;
#pragma unroll
    for (int i = 0; i < 8; ++i) q[lane + i * 32] = (_Float16)(v[i] * inv);
}

// Flash attention: one wave per (b, h, 16-row M tile). K/V f16 are L2-resident.
__global__ __launch_bounds__(256) void attn(const _Float16* __restrict__ Qn,
                                            const _Float16* __restrict__ Kp,
                                            const _Float16* __restrict__ Vp,
                                            float* __restrict__ out) {
    __shared__ _Float16 pLds[8][16 * 32];     // per-wave P tile (16m x 32n)
    const int wave = threadIdx.x >> 5;
    const int lane = threadIdx.x & 31;
    const int wid = blockIdx.x * 8 + wave;    // 0..2047
    const int mt = wid & 63;
    const int h  = (wid >> 6) & (HEADS - 1);
    const int b  = wid >> 8;
    const int m0 = mt * 16;

    // Preload Q A-fragments: [16, 256] = 8 k-steps
    const _Float16* qBase = Qn + ((size_t)(b * MROWS + m0)) * QK_DIM + h * MD;
    v16h qf[8];
#pragma unroll
    for (int i = 0; i < 8; ++i) qf[i] = load_a_frag(qBase + i * 32, QK_DIM);

    v8f o[4];
    float rmax[8], rsum[8];
#pragma unroll
    for (int t = 0; t < 4; ++t)
#pragma unroll
        for (int r = 0; r < 8; ++r) o[t][r] = 0.0f;
#pragma unroll
    for (int r = 0; r < 8; ++r) { rmax[r] = -INFINITY; rsum[r] = 0.f; }

    _Float16* pbuf = &pLds[wave][0];
    const float sc = 1.0f / (float)MD;

    for (int n0 = 0; n0 < NROWS; n0 += 32) {
        // scores: S[16m x 32n] = Q [16,256] x K^T, two 16-col WMMA tiles
        v8f s0, s1;
#pragma unroll
        for (int r = 0; r < 8; ++r) { s0[r] = 0.f; s1[r] = 0.f; }
        const _Float16* kBase = Kp + ((size_t)(b * NROWS + n0)) * QK_DIM + h * MD;
#pragma unroll
        for (int i = 0; i < 8; ++i) {
            v16h kb0 = load_bT_frag(kBase + i * 32, QK_DIM);
            v16h kb1 = load_bT_frag(kBase + (size_t)16 * QK_DIM + i * 32, QK_DIM);
            s0 = wmma_f16(qf[i], kb0, s0);
            s1 = wmma_f16(qf[i], kb1, s1);
        }

        // online softmax per row (rows live in 16-lane halves)
        float nscale[8];
#pragma unroll
        for (int r = 0; r < 8; ++r) {
            float a0 = s0[r] * sc, a1 = s1[r] * sc;
            float mx = fmaxf(a0, a1);
#pragma unroll
            for (int msk = 8; msk >= 1; msk >>= 1)
                mx = fmaxf(mx, __shfl_xor(mx, msk, 32));
            float nm = fmaxf(rmax[r], mx);
            float scale = __expf(rmax[r] - nm);
            float p0 = __expf(a0 - nm);
            float p1 = __expf(a1 - nm);
            float ps = p0 + p1;
#pragma unroll
            for (int msk = 8; msk >= 1; msk >>= 1) ps += __shfl_xor(ps, msk, 32);
            rsum[r] = rsum[r] * scale + ps;
            rmax[r] = nm;
            nscale[r] = scale;
            // C layout -> row-major LDS tile
            int rr = r + ((lane >> 4) << 3);
            pbuf[rr * 32 + (lane & 15)]      = (_Float16)p0;
            pbuf[rr * 32 + (lane & 15) + 16] = (_Float16)p1;
        }
#pragma unroll
        for (int t = 0; t < 4; ++t)
#pragma unroll
            for (int r = 0; r < 8; ++r) o[t][r] *= nscale[r];

        // P as A-fragment from LDS (re-shape C layout -> A layout)
        v16h pf = load_a_frag(pbuf, 32);

        // PV: V is row-major [N, 256]; gather B fragments per 16-col tile
        const _Float16* vBase = Vp + ((size_t)(b * NROWS + n0)) * V_DIM + h * DV;
#pragma unroll
        for (int t = 0; t < 4; ++t) {
            v16h vf = load_b_rowmajor_frag(vBase + t * 16, V_DIM);
            o[t] = wmma_f16(pf, vf, o[t]);
        }
    }

    // epilogue: normalize by softmax denominator, store fp32
    const int cl = lane & 15;
    const int rh = (lane >> 4) << 3;
#pragma unroll
    for (int t = 0; t < 4; ++t)
#pragma unroll
        for (int r = 0; r < 8; ++r) {
            int m = m0 + r + rh;
            out[((size_t)(b * MROWS + m)) * V_DIM + h * DV + t * 16 + cl] =
                o[t][r] / rsum[r];
        }
}

// ---- host launcher ----------------------------------------------------------

extern "C" void kernel_launch(void* const* d_in, const int* in_sizes, int n_in,
                              void* d_out, int out_size, void* d_ws, size_t ws_size,
                              hipStream_t stream) {
    const float* first_app  = (const float*)d_in[0];
    const float* second_app = (const float*)d_in[1];
    const float* Wk = (const float*)d_in[2];
    const float* bk = (const float*)d_in[3];
    const float* Wq = (const float*)d_in[4];
    const float* bq = (const float*)d_in[5];
    const float* Wv = (const float*)d_in[6];
    const float* bv = (const float*)d_in[7];
    float* out = (float*)d_out;

    // workspace carve-up (all sizes 256B-multiples)
    const size_t actElems = (size_t)BATCH * NROWS * IN_DIM;   // 102,760,448
    size_t off = 0;
    char* ws = (char*)d_ws;
    auto take = [&](size_t bytes) { char* p = ws + off; off += bytes; return p; };
    _Float16* A1h = (_Float16*)take(actElems * 2);                    // first_app f16
    _Float16* A2h = (_Float16*)take(actElems * 2);                    // second_app f16
    _Float16* WkT = (_Float16*)take((size_t)QK_DIM * IN_DIM * 2);
    _Float16* WqT = (_Float16*)take((size_t)QK_DIM * IN_DIM * 2);
    _Float16* WvT = (_Float16*)take((size_t)V_DIM * IN_DIM * 2);
    _Float16* Kp  = (_Float16*)take((size_t)BATCH * NROWS * QK_DIM * 2);
    _Float16* Qp  = (_Float16*)take((size_t)BATCH * MROWS * QK_DIM * 2);
    _Float16* Qn  = (_Float16*)take((size_t)BATCH * MROWS * QK_DIM * 2);
    _Float16* Vp  = (_Float16*)take((size_t)BATCH * NROWS * V_DIM * 2);
    if (ws_size < off) return;  // not enough scratch; avoid OOB

    // 1) activations fp32 -> f16
    cvt_f32_f16<<<32768, 256, 0, stream>>>(first_app,  A1h, actElems / 4);
    cvt_f32_f16<<<32768, 256, 0, stream>>>(second_app, A2h, actElems / 4);

    // 2) weights fp32 -> f16, transposed to [cols, IN_DIM]
    transpose_cvt<<<dim3(QK_DIM / 32, IN_DIM / 32), 256, 0, stream>>>(Wk, WkT, IN_DIM, QK_DIM);
    transpose_cvt<<<dim3(QK_DIM / 32, IN_DIM / 32), 256, 0, stream>>>(Wq, WqT, IN_DIM, QK_DIM);
    transpose_cvt<<<dim3(V_DIM  / 32, IN_DIM / 32), 256, 0, stream>>>(Wv, WvT, IN_DIM, V_DIM);

    // 3) projections (WMMA GEMMs): rows = B*1024 = 8192
    gemm_proj<<<dim3(64, QK_DIM / 128), 256, 0, stream>>>(A1h, WkT, bk, Kp, QK_DIM);
    gemm_proj<<<dim3(64, QK_DIM / 128), 256, 0, stream>>>(A2h, WqT, bq, Qp, QK_DIM);
    gemm_proj<<<dim3(64, V_DIM  / 128), 256, 0, stream>>>(A1h, WvT, bv, Vp, V_DIM);

    // 4) per-head L2 normalization of Q: 8192*4 waves / 8 per block
    qnorm<<<4096, 256, 0, stream>>>(Qp, Qn);

    // 5) flash attention: 8*4*64 = 2048 waves, 8 waves/block
    attn<<<256, 256, 0, stream>>>(Qn, Kp, Vp, out);
}
